// MultiHeadAttention_24000277250482
// MI455X (gfx1250) — matrix operand order repaired and checked by reading, not compiled, not run on hardware
//
#include <hip/hip_runtime.h>

// ---------------- problem constants ----------------
#define BATCH 4
#define SEQ   2048
#define DMODEL 1024
#define NHEAD 16
#define DHEAD 64
#define NEGBIG 1000000000.0f
#define LOG2E 1.44269504088896340736f

typedef __attribute__((ext_vector_type(16))) __bf16 v16bf;
typedef __attribute__((ext_vector_type(8)))  float  v8f;
typedef __attribute__((ext_vector_type(4)))  unsigned int v4u;
typedef __attribute__((ext_vector_type(8)))  int v8i;
typedef __attribute__((ext_vector_type(4)))  int v4i;

// ---------------- CDNA5 data-mover feature detection ----------------
#if defined(__HIP_DEVICE_COMPILE__)
  #if __has_builtin(__builtin_amdgcn_tensor_load_to_lds)
    #if __has_include(<hip/amd_detail/amd_gfx1250_TDM.h>)
      #define TDM_6ARG 1
      #pragma message("CDNA5 path: tensor_load_to_lds (6-arg, therock toolchain)")
    #else
      #define TDM_5ARG 1
      #pragma message("CDNA5 path: tensor_load_to_lds (5-arg, ROCm 7.2 toolchain)")
    #endif
  #elif __has_builtin(__builtin_amdgcn_global_load_async_to_lds_b128)
    #define USE_ASYNC 1
    #pragma message("CDNA5 path: global_load_async_to_lds_b128")
  #else
    #pragma message("CDNA5 path: plain LDS staging (no TDM/async builtins found)")
  #endif
  #if __has_builtin(__builtin_amdgcn_s_wait_asynccnt)
    #define WAIT_ASYNC() __builtin_amdgcn_s_wait_asynccnt(0)
  #else
    #define WAIT_ASYNC() asm volatile("s_wait_asynccnt 0x0" ::: "memory")
  #endif
#endif

#if defined(TDM_5ARG) || defined(TDM_6ARG)
#define USE_TDM 1
#endif

union FragBf {
    uint4 u[2];
    v16bf v;
};

static __device__ __forceinline__ unsigned short f32_to_bf16(float f) {
    unsigned int u = __float_as_uint(f);
    u += 0x7FFFu + ((u >> 16) & 1u);   // round-to-nearest-even
    return (unsigned short)(u >> 16);
}

static __device__ __forceinline__ v8f wmma_bf16(const FragBf& a, const FragBf& b, v8f c) {
    return __builtin_amdgcn_wmma_f32_16x16x32_bf16(
        /*neg_a=*/false, a.v, /*neg_b=*/false, b.v,
        /*c_mod=*/(short)0, c, /*reuse_a=*/false, /*reuse_b=*/false);
}

#if defined(USE_TDM)
// Absolute LDS byte offset of a __shared__ pointer (generic -> AS3 -> int)
static __device__ __forceinline__ unsigned lds_offset_of(const void* p) {
    return (unsigned)(unsigned long long)(__attribute__((address_space(3))) const void*)p;
}

// Issue a TDM 2-D tile load: 32 rows x 64 bf16 from global (row stride 64
// elems) into LDS with +16B padding after every 128B row (LDS stride 144B).
static __device__ __forceinline__ void tdm_load_k_tile(const unsigned short* gsrc,
                                                       unsigned lds_byte_off) {
    unsigned long long ga = (unsigned long long)(const void*)gsrc;
    v4u g0 = {};
    g0.x = 1u;                              // count=1 (valid), user mode, no gather
    g0.y = lds_byte_off;                    // lds_addr (bytes)
    g0.z = (unsigned)(ga & 0xFFFFFFFFull);  // global_addr[31:0]
    g0.w = (unsigned)((ga >> 32) & 0x1FFFFFFull) | (2u << 30);  // addr[56:32] | type=2

    v8i g1 = {};
    // data_size=1 (2B), pad_enable, pad_interval=4 (32 DWORDs = 128B row),
    // pad_amount=3 (4 DWORDs = 16B)
    g1[0] = (1 << 16) | (1 << 20) | (4 << 22) | (3 << 25);
    g1[1] = (64 << 16);          // tensor_dim0[15:0]=64 in bits[31:16]
    g1[2] = (32 << 16);          // tensor_dim0 hi=0 ; tensor_dim1[15:0]=32
    g1[3] = (64 << 16);          // tensor_dim1 hi=0 ; tile_dim0=64
    g1[4] = 32;                  // tile_dim1=32 ; tile_dim2=0
    g1[5] = 64;                  // tensor_dim0_stride lo = 64 elements
    g1[6] = 0;                   // stride hi / tensor_dim1_stride (unused, 2D)
    g1[7] = 0;

    v4i g2 = {};                 // unused (2-D tensor)
    v4i g3 = {};
#if defined(TDM_6ARG)
    v8i g4 = {};
    __builtin_amdgcn_tensor_load_to_lds(g0, g1, g2, g3, g4, 0);
#else
    __builtin_amdgcn_tensor_load_to_lds(g0, g1, g2, g3, 0);
#endif
}
#endif

// =====================================================================
// Projection GEMM:  C[m,n] = sum_k act[m,k] * W[k,n]
//   act: [B*S, 1024] f32, W: [1024, 1024] f32
//   out: bf16 in [B, H, S, 64] layout
// Workgroup tile 128x128, 8 waves (each 32x64), K-step 32.
// =====================================================================
__global__ __launch_bounds__(256) void proj_kernel(
    const float* __restrict__ act, const float* __restrict__ W,
    unsigned short* __restrict__ out)
{
    __shared__ unsigned short lA[128 * 40];   // [row][k] stride 40 (80B, 16B aligned)
    __shared__ unsigned short lB[128 * 40];   // [n][k] transposed, stride 40

    const int tid  = threadIdx.x;
    const int lane = tid & 31;
    const int wave = tid >> 5;
    const int m0 = blockIdx.x * 128;
    const int n0 = blockIdx.y * 128;
    const int wm = wave & 3;      // 4 m-groups of 32 rows
    const int wn = wave >> 2;     // 2 n-groups of 64 cols

    const int l16   = lane & 15;
    const int lhalf = lane >> 4;
    const int kb8   = lhalf * 8;    // A-fragment K base (0 or 8)
    const int ko16  = lhalf * 16;   // unused (B fragments also use kb8)

    v8f acc[2][4] = {};

    for (int kk = 0; kk < DMODEL; kk += 32) {
        __syncthreads();
        // ---- stage A tile 128x32 f32 -> bf16 LDS ----
        for (int j = 0; j < 4; ++j) {
            int slot = tid + j * 256;          // 1024 float4 slots
            int r  = slot >> 3;
            int c4 = slot & 7;
            float4 f = *(const float4*)(act + (size_t)(m0 + r) * DMODEL + kk + c4 * 4);
            uint2 pk;
            pk.x = (unsigned)f32_to_bf16(f.x) | ((unsigned)f32_to_bf16(f.y) << 16);
            pk.y = (unsigned)f32_to_bf16(f.z) | ((unsigned)f32_to_bf16(f.w) << 16);
            *(uint2*)&lA[r * 40 + c4 * 4] = pk;
        }
        // ---- stage B tile 32x128 f32, transposed -> [n][k] bf16 LDS ----
        for (int j = 0; j < 4; ++j) {
            int slot = tid + j * 256;          // 1024 float4 slots
            int r  = slot >> 5;                // k row 0..31
            int c4 = slot & 31;                // n vec
            float4 f = *(const float4*)(W + (size_t)(kk + r) * DMODEL + n0 + c4 * 4);
            lB[(c4 * 4 + 0) * 40 + r] = f32_to_bf16(f.x);
            lB[(c4 * 4 + 1) * 40 + r] = f32_to_bf16(f.y);
            lB[(c4 * 4 + 2) * 40 + r] = f32_to_bf16(f.z);
            lB[(c4 * 4 + 3) * 40 + r] = f32_to_bf16(f.w);
        }
        __syncthreads();

        if (kk + 32 < DMODEL) {   // prefetch next tiles (global_prefetch_b8)
            __builtin_prefetch(act + (size_t)(m0 + (tid >> 1)) * DMODEL + kk + 32, 0, 0);
            __builtin_prefetch(W + (size_t)(kk + 32 + (tid >> 3)) * DMODEL + n0, 0, 0);
        }

        FragBf a[2], b[4];
        for (int tm = 0; tm < 2; ++tm) {
            int row = (wm * 2 + tm) * 16 + l16;
            a[tm].u[0] = *(const uint4*)&lA[row * 40 + kb8];
            a[tm].u[1] = *(const uint4*)&lA[row * 40 + kb8 + 16];
        }
        for (int nt = 0; nt < 4; ++nt) {
            int nr = wn * 64 + nt * 16 + l16;
            b[nt].u[0] = *(const uint4*)&lB[nr * 40 + kb8];
            b[nt].u[1] = *(const uint4*)&lB[nr * 40 + kb8 + 16];
        }
        for (int tm = 0; tm < 2; ++tm)
            for (int nt = 0; nt < 4; ++nt)
                acc[tm][nt] = wmma_bf16(a[tm], b[nt], acc[tm][nt]);
    }

    // ---- store bf16 into [B, H, S, 64] ----
    for (int tm = 0; tm < 2; ++tm) {
        for (int nt = 0; nt < 4; ++nt) {
            for (int r = 0; r < 8; ++r) {
                int m = m0 + (wm * 2 + tm) * 16 + r + lhalf * 8;
                int n = n0 + wn * 64 + nt * 16 + l16;
                int bb = m >> 11, s = m & (SEQ - 1);
                int h  = n >> 6,  d = n & (DHEAD - 1);
                out[(((size_t)bb * NHEAD + h) * SEQ + s) * DHEAD + d] =
                    f32_to_bf16(acc[tm][nt][r]);
            }
        }
    }
}

// =====================================================================
// Flash attention: one WG = 4 waves = 64 query rows of one (b,h).
// k-loop in 32-key steps, causal-bounded. bf16 WMMA, f32 online softmax.
// K tile staged via TDM / async-to-LDS when available.
// =====================================================================
__global__ __launch_bounds__(128) void attn_kernel(
    const unsigned short* __restrict__ qp, const unsigned short* __restrict__ kp,
    const unsigned short* __restrict__ vp, const float* __restrict__ v_mask,
    const float* __restrict__ q_mask, float* __restrict__ out)
{
    __shared__ unsigned short lK[32 * 72];        // [key][d]   stride 72 (144B)
    __shared__ unsigned short lV[64 * 40];        // [d][key]   transposed, stride 40
    __shared__ unsigned short lP[4 * 16 * 40];    // per-wave P tile [16 q][32 k] stride 40

    const int bh = blockIdx.x;            // 0..63
    const int b  = bh >> 4;
    const int h  = bh & (NHEAD - 1);
    const int qb = blockIdx.y * 64;
    const int tid  = threadIdx.x;
    const int lane = tid & 31;
    const int wave = tid >> 5;
    const int l16   = lane & 15;
    const int lhalf = lane >> 4;
    const int kb8   = lhalf * 8;
    const int ko16  = lhalf * 16;

    const size_t base = (size_t)bh * SEQ * DHEAD;

    // ---- preload Q A-fragments (16 rows x 64 d) ----
    const int qrow = qb + wave * 16 + l16;
    const unsigned short* qrp = qp + base + (size_t)qrow * DHEAD;
    FragBf qf[2];
    qf[0].u[0] = *(const uint4*)(qrp + kb8);
    qf[0].u[1] = *(const uint4*)(qrp + kb8 + 16);
    qf[1].u[0] = *(const uint4*)(qrp + 32 + kb8);
    qf[1].u[1] = *(const uint4*)(qrp + 48 + kb8);

    float mrow[8], lrow[8];
    v8f acco[4] = {};
    for (int r = 0; r < 8; ++r) { mrow[r] = -3.0e38f; lrow[r] = 0.0f; }

    unsigned short* Pw = &lP[wave * 16 * 40];
    const int nsteps = 2 * (blockIdx.y + 1);   // causal bound: keys < qb+64

#if defined(USE_TDM)
    const unsigned lK_off = lds_offset_of(&lK[0]);
#endif

    for (int st = 0; st < nsteps; ++st) {
        const int kbase = st * 32;
        __syncthreads();
#if defined(USE_TDM)
        // ---- stage K tile [32][64] bf16 via Tensor Data Mover ----
        if (wave == 0) {
            tdm_load_k_tile(kp + base + (size_t)kbase * DHEAD, lK_off);
        }
#elif defined(USE_ASYNC)
        // ---- stage K tile via async global->LDS (ASYNCcnt) ----
        for (int j = 0; j < 2; ++j) {
            int slot = tid + j * 128;          // 256 uint4 slots
            int r = slot >> 3, c = slot & 7;
            const unsigned short* g = kp + base + (size_t)(kbase + r) * DHEAD + c * 8;
            __builtin_amdgcn_global_load_async_to_lds_b128(
                (const __attribute__((address_space(1))) void*)g,
                (__attribute__((address_space(3))) void*)&lK[r * 72 + c * 8],
                0, 0);
        }
#else
        // ---- stage K tile [32][64] bf16 (vector copy) ----
        for (int j = 0; j < 2; ++j) {
            int slot = tid + j * 128;          // 256 uint4 slots
            int r = slot >> 3, c = slot & 7;
            uint4 d = *(const uint4*)(kp + base + (size_t)(kbase + r) * DHEAD + c * 8);
            *(uint4*)&lK[r * 72 + c * 8] = d;
        }
#endif
        // ---- stage V tile transposed -> [d][key] ----
        for (int j = 0; j < 2; ++j) {
            int slot = tid + j * 128;
            int r = slot >> 3, c = slot & 7;
            uint4 d = *(const uint4*)(vp + base + (size_t)(kbase + r) * DHEAD + c * 8);
            unsigned short e[8];
            *(uint4*)e = d;
            for (int i = 0; i < 8; ++i) lV[(c * 8 + i) * 40 + r] = e[i];
        }
#if defined(USE_TDM)
        if (wave == 0) __builtin_amdgcn_s_wait_tensorcnt(0);
#elif defined(USE_ASYNC)
        WAIT_ASYNC();
#endif
        __syncthreads();

        // ---- scores: two 16-key sub-tiles ----
        v8f sc[2];
        for (int kt = 0; kt < 2; ++kt) {
            FragBf kf0, kf1;
            int krow = kt * 16 + l16;
            kf0.u[0] = *(const uint4*)&lK[krow * 72 + kb8];
            kf0.u[1] = *(const uint4*)&lK[krow * 72 + kb8 + 16];
            kf1.u[0] = *(const uint4*)&lK[krow * 72 + 32 + kb8];
            kf1.u[1] = *(const uint4*)&lK[krow * 72 + 32 + kb8 + 16];
            v8f c = {};
            c = wmma_bf16(qf[0], kf0, c);      // d 0..31
            c = wmma_bf16(qf[1], kf1, c);      // d 32..63
            int key   = kbase + kt * 16 + l16;
            float vm  = v_mask[b * SEQ + key];
            float add = (1.0f - vm) * -NEGBIG;
            for (int r = 0; r < 8; ++r) {
                int qg = qb + wave * 16 + r + lhalf * 8;
                float s = c[r] * 0.125f + add + (key > qg ? -NEGBIG : 0.0f);
                sc[kt][r] = s;
            }
        }

        // ---- online softmax (row = 16 lanes of one half-wave) ----
        float corr[8];
        for (int r = 0; r < 8; ++r) {
            float rm = fmaxf(sc[0][r], sc[1][r]);
            for (int off = 1; off < 16; off <<= 1)
                rm = fmaxf(rm, __shfl_xor(rm, off, 32));
            float mn = fmaxf(mrow[r], rm);
            corr[r]  = exp2f((mrow[r] - mn) * LOG2E);
            mrow[r]  = mn;
        }
        for (int kt = 0; kt < 2; ++kt) {
            for (int r = 0; r < 8; ++r) {
                float p = exp2f((sc[kt][r] - mrow[r]) * LOG2E);
                sc[kt][r] = p;
                int prow = r + lhalf * 8;
                Pw[prow * 40 + kt * 16 + l16] = f32_to_bf16(p);
            }
        }
        for (int r = 0; r < 8; ++r) {
            float rs = sc[0][r] + sc[1][r];
            for (int off = 1; off < 16; off <<= 1)
                rs += __shfl_xor(rs, off, 32);
            lrow[r] = lrow[r] * corr[r] + rs;
        }
        for (int dt = 0; dt < 4; ++dt)
            for (int r = 0; r < 8; ++r)
                acco[dt][r] *= corr[r];

        __syncthreads();   // P stores (cross-lane) visible before A-fragment loads

        // ---- O += P (16x32) x V^T tiles ----
        FragBf pf;
        pf.u[0] = *(const uint4*)&Pw[l16 * 40 + kb8];
        pf.u[1] = *(const uint4*)&Pw[l16 * 40 + kb8 + 16];
        for (int dt = 0; dt < 4; ++dt) {
            FragBf vf;
            int vrow = dt * 16 + l16;
            vf.u[0] = *(const uint4*)&lV[vrow * 40 + kb8];
            vf.u[1] = *(const uint4*)&lV[vrow * 40 + kb8 + 16];
            acco[dt] = wmma_bf16(pf, vf, acco[dt]);
        }
    }

    // ---- epilogue: /l, *q_mask, store f32 [B, S, H*64] ----
    for (int r = 0; r < 8; ++r) {
        int qg = qb + wave * 16 + r + lhalf * 8;
        float qm  = q_mask[b * SEQ + qg];
        float inv = qm / lrow[r];
        for (int dt = 0; dt < 4; ++dt) {
            int d = dt * 16 + l16;
            out[((size_t)(b * SEQ + qg)) * (NHEAD * DHEAD) + h * DHEAD + d] =
                acco[dt][r] * inv;
        }
    }
}

// =====================================================================
extern "C" void kernel_launch(void* const* d_in, const int* in_sizes, int n_in,
                              void* d_out, int out_size, void* d_ws, size_t ws_size,
                              hipStream_t stream) {
    (void)in_sizes; (void)n_in; (void)out_size; (void)ws_size;
    const float* q      = (const float*)d_in[0];
    const float* k      = (const float*)d_in[1];
    const float* v      = (const float*)d_in[2];
    const float* v_mask = (const float*)d_in[3];
    const float* q_mask = (const float*)d_in[4];
    const float* Wq     = (const float*)d_in[5];
    const float* Wk     = (const float*)d_in[6];
    const float* Wv     = (const float*)d_in[7];

    const size_t proj_elems = (size_t)BATCH * NHEAD * SEQ * DHEAD;  // 8,388,608
    unsigned short* qp = (unsigned short*)d_ws;
    unsigned short* kp = qp + proj_elems;
    unsigned short* vp = kp + proj_elems;

    dim3 pgrid(BATCH * SEQ / 128, DMODEL / 128);   // 64 x 8
    proj_kernel<<<pgrid, 256, 0, stream>>>(q, Wq, qp);
    proj_kernel<<<pgrid, 256, 0, stream>>>(k, Wk, kp);
    proj_kernel<<<pgrid, 256, 0, stream>>>(v, Wv, vp);

    dim3 agrid(BATCH * NHEAD, SEQ / 64);           // 64 x 32
    attn_kernel<<<agrid, 128, 0, stream>>>(qp, kp, vp, v_mask, q_mask, (float*)d_out);
}
